// EvolveGCN_66262755443074
// MI455X (gfx1250) — matrix operand order repaired and checked by reading, not compiled, not run on hardware
//
#include <hip/hip_runtime.h>

// ---------------------------------------------------------------------------
// Problem constants (from reference)
// ---------------------------------------------------------------------------
namespace {
constexpr int T_    = 8;
constexpr int B_    = 16;
constexpr int N_    = 400;
constexpr int E_    = 12800;
constexpr int FIN_  = 64;
constexpr int FH_   = 64;
constexpr int FOUT_ = 32;
constexpr int H1_   = FIN_ * FH_;    // 4096
constexpr int H2_   = FH_ * FOUT_;   // 2048
}

typedef __attribute__((ext_vector_type(16))) __bf16 v16bf;
typedef __attribute__((ext_vector_type(8)))  float  v8f;

#define WMMA_BF16(A, B, C) \
  __builtin_amdgcn_wmma_f32_16x16x32_bf16(false, (A), false, (B), (short)0, (C), false, false)

// ---------------------------------------------------------------------------
// Fragment loaders (layouts per CDNA5 ISA 7.12.2, wave32)
//   A (16x32 bf16): lane m=lane&15; lanes 0-15 hold K [0..7],[16..23];
//                   lanes 16-31 hold K [8..15],[24..31]
//   B (32x16 bf16): lane n=lane&15; lanes 0-15 hold K [0..7],[16..23];
//                   lanes 16-31 hold K [8..15],[24..31]
//   C/D (16x16 f32): lane n=lane&15; VGPR i -> M=(lane>>4)*8+i
// ---------------------------------------------------------------------------
__device__ __forceinline__ v16bf load_frag_a_bf16(const __bf16* __restrict__ base,
                                                  int row_stride, int lane) {
  const int r  = lane & 15;
  const int kb = (lane >> 4) * 8;
  const __bf16* p = base + (size_t)r * row_stride + kb;
  union { v16bf v; uint4 q[2]; } u;
  u.q[0] = *(const uint4*)(p);
  u.q[1] = *(const uint4*)(p + 16);
  return u.v;
}

// A fragment when the per-lane pointer is maintained by the caller
// (points at element [lane&15][kb] of the current k-tile; +16 gives K+16)
__device__ __forceinline__ v16bf load_frag_a_ptr(const __bf16* __restrict__ p) {
  union { v16bf v; uint4 q[2]; } u;
  u.q[0] = *(const uint4*)(p);
  u.q[1] = *(const uint4*)(p + 16);
  return u.v;
}

__device__ __forceinline__ v16bf load_frag_a_f32(const float* __restrict__ base,
                                                 int row_stride, int lane) {
  const int r  = lane & 15;
  const int kb = (lane >> 4) * 8;
  const float* p = base + (size_t)r * row_stride + kb;
  float4 f0 = *(const float4*)(p);
  float4 f1 = *(const float4*)(p + 4);
  float4 f2 = *(const float4*)(p + 16);
  float4 f3 = *(const float4*)(p + 20);
  union { v16bf v; __bf16 e[16]; } u;
  u.e[0]  = (__bf16)f0.x; u.e[1]  = (__bf16)f0.y; u.e[2]  = (__bf16)f0.z; u.e[3]  = (__bf16)f0.w;
  u.e[4]  = (__bf16)f1.x; u.e[5]  = (__bf16)f1.y; u.e[6]  = (__bf16)f1.z; u.e[7]  = (__bf16)f1.w;
  u.e[8]  = (__bf16)f2.x; u.e[9]  = (__bf16)f2.y; u.e[10] = (__bf16)f2.z; u.e[11] = (__bf16)f2.w;
  u.e[12] = (__bf16)f3.x; u.e[13] = (__bf16)f3.y; u.e[14] = (__bf16)f3.z; u.e[15] = (__bf16)f3.w;
  return u.v;
}

__device__ __forceinline__ v16bf load_frag_b_bf16(const __bf16* __restrict__ src,
                                                  int src_stride, int lane) {
  const int c  = lane & 15;
  const int kb = (lane >> 4) * 8;
  const __bf16* p = src + (size_t)c * src_stride + kb;
  union { v16bf v; uint4 q[2]; } u;
  u.q[0] = *(const uint4*)(p);
  u.q[1] = *(const uint4*)(p + 16);
  return u.v;
}

// Pre-packed B fragment: 32 contiguous bytes per lane at p (p already +lane*16)
__device__ __forceinline__ v16bf load_frag_packed(const __bf16* __restrict__ p) {
  union { v16bf v; uint4 q[2]; } u;
  u.q[0] = *(const uint4*)(p);
  u.q[1] = *(const uint4*)(p + 8);
  return u.v;
}

// f32 runs p[0..7], p[16..23] -> bf16 fragment half-pair (pointer at row/k offset)
__device__ __forceinline__ v16bf cvt16_f32(const float* __restrict__ p) {
  float4 f0 = *(const float4*)(p);
  float4 f1 = *(const float4*)(p + 4);
  float4 f2 = *(const float4*)(p + 16);
  float4 f3 = *(const float4*)(p + 20);
  union { v16bf v; __bf16 e[16]; } u;
  u.e[0]  = (__bf16)f0.x; u.e[1]  = (__bf16)f0.y; u.e[2]  = (__bf16)f0.z; u.e[3]  = (__bf16)f0.w;
  u.e[4]  = (__bf16)f1.x; u.e[5]  = (__bf16)f1.y; u.e[6]  = (__bf16)f1.z; u.e[7]  = (__bf16)f1.w;
  u.e[8]  = (__bf16)f2.x; u.e[9]  = (__bf16)f2.y; u.e[10] = (__bf16)f2.z; u.e[11] = (__bf16)f2.w;
  u.e[12] = (__bf16)f3.x; u.e[13] = (__bf16)f3.y; u.e[14] = (__bf16)f3.z; u.e[15] = (__bf16)f3.w;
  return u.v;
}

// ---------------------------------------------------------------------------
// Step 0 of the GRU chain needs no matmul (h==0): h = f(bih, bhh) only.
// Writes the same H-vector to all 16 batch rows.
// ---------------------------------------------------------------------------
__global__ void __launch_bounds__(256) gru_init_kernel(
    const float* __restrict__ bih, const float* __restrict__ bhh,
    __bf16* __restrict__ h_out, int H) {
  const int j = blockIdx.x * 256 + threadIdx.x;
  const float r = 1.f / (1.f + expf(-(bih[j] + bhh[j])));
  const float z = 1.f / (1.f + expf(-(bih[H + j] + bhh[H + j])));
  const float n = tanhf(bih[2 * H + j] + r * bhh[2 * H + j]);
  const __bf16 hb = (__bf16)((1.f - z) * n);
  for (int b = 0; b < B_; ++b) h_out[(size_t)b * H + j] = hb;
}

// ---------------------------------------------------------------------------
// Pack Wih (f32, [3H][H]) into bf16 B-fragments, one 1 KB block per
// (row-tile, k-tile): wpack[((tl*ktiles)+kt)*512 + lane*16 + i]
//   tl = g*(H/16)+jt  ->  W row = tl*16 + (lane&15)
//   k  = kt*32 + (lane>>4)*8 + (i&7) + (i>>3)*16
// After this pass the full packed weights (125.9 MB for both layers) fit in
// the 192 MB L2, so the 7 matmul steps stream from L2 instead of HBM.
// ---------------------------------------------------------------------------
__global__ void __launch_bounds__(256) pack_w_kernel(
    const float* __restrict__ W, __bf16* __restrict__ wpack, int H) {
  const int idx = blockIdx.x * 256 + threadIdx.x;   // 3H/16 * H/32 * 512 threads
  const int i      = idx & 15;
  const int lane   = (idx >> 4) & 31;
  const int rem    = idx >> 9;
  const int ktiles = H >> 5;
  const int kt     = rem % ktiles;
  const int tl     = rem / ktiles;
  const int row    = tl * 16 + (lane & 15);
  const int col    = kt * 32 + ((lane >> 4) << 3) + (i & 7) + ((i >> 3) << 4);
  wpack[idx] = (__bf16)W[(size_t)row * H + col];
}

// ---------------------------------------------------------------------------
// GRU step (packed-weight fast path): one workgroup per 16-column tile of H,
// 16 waves split K, LDS reduce, gates.  All four fragments are issued before
// the three WMMAs so load latency overlaps matrix work (partial loadcnt
// waits instead of draining to zero), and per-lane pointers are simply
// incremented (no per-iteration address rebuild).
// ---------------------------------------------------------------------------
__global__ void __launch_bounds__(512) gru_step_packed_kernel(
    const __bf16* __restrict__ h_in,
    const __bf16* __restrict__ wpack,
    const float*  __restrict__ bih,
    const float*  __restrict__ bhh,
    __bf16*       __restrict__ h_out,
    int H) {
  __shared__ float part[16][3][256];
  const int tid  = threadIdx.x;
  const int lane = tid & 31;
  const int wave = tid >> 5;
  const int jt   = blockIdx.x;
  const int K    = H;
  const int ktiles   = K >> 5;        // k-tiles per gate row-tile
  const int kt_per_w = ktiles >> 4;   // 8 (H=4096) or 4 (H=2048)
  const int kt0      = wave * kt_per_w;
  const int HT       = H >> 4;        // row-tiles per gate

  const __bf16* p0 = wpack + ((size_t)(0 * HT + jt) * ktiles + kt0) * 512 + lane * 16;
  const __bf16* p1 = wpack + ((size_t)(1 * HT + jt) * ktiles + kt0) * 512 + lane * 16;
  const __bf16* p2 = wpack + ((size_t)(2 * HT + jt) * ktiles + kt0) * 512 + lane * 16;
  const __bf16* pa = h_in + (size_t)(lane & 15) * K + ((lane >> 4) * 8) + (kt0 << 5);

  v8f acc0 = {}, acc1 = {}, acc2 = {};
#pragma unroll 4
  for (int i = 0; i < kt_per_w; ++i) {
    // Issue all loads first; WMMAs consume them with partial waits.
    v16bf a  = load_frag_a_ptr(pa);
    v16bf b0 = load_frag_packed(p0);
    v16bf b1 = load_frag_packed(p1);
    v16bf b2 = load_frag_packed(p2);
    __builtin_prefetch(p0 + 2 * 512, 0, 1);
    __builtin_prefetch(p1 + 2 * 512, 0, 1);
    __builtin_prefetch(p2 + 2 * 512, 0, 1);
    acc0 = WMMA_BF16(a, b0, acc0);
    acc1 = WMMA_BF16(a, b1, acc1);
    acc2 = WMMA_BF16(a, b2, acc2);
    pa += 32; p0 += 512; p1 += 512; p2 += 512;
  }

  const int c  = lane & 15;
  const int mb = (lane >> 4) * 8;
#pragma unroll
  for (int i = 0; i < 8; ++i) {
    part[wave][0][(mb + i) * 16 + c] = acc0[i];
    part[wave][1][(mb + i) * 16 + c] = acc1[i];
    part[wave][2][(mb + i) * 16 + c] = acc2[i];
  }
  __syncthreads();
  if (tid < 256) {
    float s0 = 0.f, s1 = 0.f, s2 = 0.f;
#pragma unroll
    for (int w = 0; w < 16; ++w) {
      s0 += part[w][0][tid];
      s1 += part[w][1][tid];
      s2 += part[w][2][tid];
    }
    const int b  = tid >> 4;
    const int jj = tid & 15;
    const int j  = jt * 16 + jj;
    const float r = 1.f / (1.f + expf(-(s0 + bih[j] + bhh[j])));
    const float z = 1.f / (1.f + expf(-(s1 + bih[H + j] + bhh[H + j])));
    const float n = tanhf(s2 + bih[2 * H + j] + r * bhh[2 * H + j]);
    h_out[(size_t)b * H + j] = (__bf16)((1.f - z) * n);
  }
}

// ---------------------------------------------------------------------------
// GRU step (fallback: convert W f32->bf16 on the fly; used if ws too small)
// ---------------------------------------------------------------------------
__global__ void __launch_bounds__(256) gru_step_kernel(
    const __bf16* __restrict__ h_in,
    const float*  __restrict__ W,
    const float*  __restrict__ bih,
    const float*  __restrict__ bhh,
    __bf16*       __restrict__ h_out,
    int H) {
  __shared__ float part[8][3][256];
  const int tid  = threadIdx.x;
  const int lane = tid & 31;
  const int wave = tid >> 5;
  const int jt   = blockIdx.x;
  const int K    = H;
  const int kslice = K >> 3;
  const int k0   = wave * kslice;
  const int c    = lane & 15;
  const int kbB  = (lane >> 4) << 3;

  const size_t rowj = (size_t)jt * 16 + c;
  const float* w0 = W + (rowj)                 * (size_t)K + kbB;
  const float* w1 = W + (rowj + (size_t)H)     * (size_t)K + kbB;
  const float* w2 = W + (rowj + 2 * (size_t)H) * (size_t)K + kbB;

  v8f acc0 = {}, acc1 = {}, acc2 = {};
  for (int kt = k0; kt < k0 + kslice; kt += 32) {
    v16bf a = load_frag_a_bf16(h_in + kt, K, lane);
    { v16bf b = cvt16_f32(w0 + kt); acc0 = WMMA_BF16(a, b, acc0); }
    { v16bf b = cvt16_f32(w1 + kt); acc1 = WMMA_BF16(a, b, acc1); }
    { v16bf b = cvt16_f32(w2 + kt); acc2 = WMMA_BF16(a, b, acc2); }
  }

  const int mb = (lane >> 4) * 8;
#pragma unroll
  for (int i = 0; i < 8; ++i) {
    part[wave][0][(mb + i) * 16 + c] = acc0[i];
    part[wave][1][(mb + i) * 16 + c] = acc1[i];
    part[wave][2][(mb + i) * 16 + c] = acc2[i];
  }
  __syncthreads();
  float s0 = 0.f, s1 = 0.f, s2 = 0.f;
#pragma unroll
  for (int w = 0; w < 8; ++w) {
    s0 += part[w][0][tid];
    s1 += part[w][1][tid];
    s2 += part[w][2][tid];
  }
  const int b  = tid >> 4;
  const int jj = tid & 15;
  const int j  = jt * 16 + jj;
  const float r = 1.f / (1.f + expf(-(s0 + bih[j] + bhh[j])));
  const float z = 1.f / (1.f + expf(-(s1 + bih[H + j] + bhh[H + j])));
  const float n = tanhf(s2 + bih[2 * H + j] + r * bhh[2 * H + j]);
  h_out[(size_t)b * H + j] = (__bf16)((1.f - z) * n);
}

// ---------------------------------------------------------------------------
// Per-batch transposed evolved weights:
//   w1t[b][fh][fin]  = h1[b][fin*FH + fh]
//   w2t[b][fout][fh] = h2[b][fh*FOUT + fout]
// ---------------------------------------------------------------------------
__global__ void __launch_bounds__(256) transposeW_kernel(
    const __bf16* __restrict__ h1, const __bf16* __restrict__ h2,
    __bf16* __restrict__ w1t, __bf16* __restrict__ w2t) {
  const int idx = blockIdx.x * 256 + threadIdx.x;
  if (idx < B_ * H1_) {
    const int b = idx / H1_, r = idx % H1_;
    const int fin = r / FH_, fh = r % FH_;
    w1t[(size_t)b * FIN_ * FH_ + (size_t)fh * FIN_ + fin] = h1[idx];
  } else {
    const int j = idx - B_ * H1_;
    const int b = j / H2_, r = j % H2_;
    const int fh = r / FOUT_, fo = r % FOUT_;
    w2t[(size_t)b * FOUT_ * FH_ + (size_t)fo * FH_ + fh] = h2[j];
  }
}

// ---------------------------------------------------------------------------
// Degree + edge norm (same edge set for both GCN layers at t=7)
// ---------------------------------------------------------------------------
__global__ void __launch_bounds__(256) deg_kernel(const int* __restrict__ ei7,
                                                  float* __restrict__ deg) {
  const int idx = blockIdx.x * 256 + threadIdx.x;
  const int b = idx / E_, e = idx % E_;
  const int row = ei7[(size_t)b * 2 * E_ + e];
  atomicAdd(deg + b * N_ + row, 1.0f);
}

__global__ void __launch_bounds__(256) norm_kernel(const int* __restrict__ ei7,
                                                   const float* __restrict__ ew7,
                                                   const float* __restrict__ deg,
                                                   float* __restrict__ norm) {
  const int idx = blockIdx.x * 256 + threadIdx.x;
  const int b = idx / E_, e = idx % E_;
  const int row = ei7[(size_t)b * 2 * E_ + e];
  const int col = ei7[(size_t)b * 2 * E_ + E_ + e];
  const float dr = deg[b * N_ + row];
  const float dc = deg[b * N_ + col];
  const float ir = dr > 0.f ? rsqrtf(dr) : 0.f;
  const float ic = dc > 0.f ? rsqrtf(dc) : 0.f;
  norm[idx] = ir * ew7[idx] * ic;
}

// support1 = x7 @ W1 : [400,64]x[64,64] per batch; grid (25,16), block 128
__global__ void __launch_bounds__(128) support1_kernel(
    const float* __restrict__ x7, const __bf16* __restrict__ w1t,
    float* __restrict__ sup1) {
  const int lane = threadIdx.x & 31;
  const int nt   = threadIdx.x >> 5;
  const int mt   = blockIdx.x;
  const int b    = blockIdx.y;
  const float*  Ab = x7  + (size_t)(b * N_ + mt * 16) * FIN_;
  const __bf16* Bt = w1t + (size_t)b * FIN_ * FH_ + (size_t)(nt * 16) * FIN_;
  v8f acc = {};
#pragma unroll
  for (int kt = 0; kt < FIN_; kt += 32) {
    v16bf a  = load_frag_a_f32(Ab + kt, FIN_, lane);
    v16bf bb = load_frag_b_bf16(Bt + kt, FIN_, lane);
    acc = WMMA_BF16(a, bb, acc);
  }
  const int n = lane & 15, mb = (lane >> 4) * 8;
  float* Cp = sup1 + (size_t)(b * N_ + mt * 16) * FH_ + nt * 16 + n;
#pragma unroll
  for (int i = 0; i < 8; ++i) Cp[(size_t)(mb + i) * FH_] = acc[i];
}

// scatter: out[row] += norm[e] * sup[col]  (x4 features per thread)
__global__ void __launch_bounds__(256) scatter_kernel(
    const float* __restrict__ sup, const float* __restrict__ norm,
    const int* __restrict__ ei7, float* __restrict__ outp, int F) {
  const int idx = blockIdx.x * 256 + threadIdx.x;
  const int fv = F >> 2;
  const int f4 = idx % fv;
  const int t  = idx / fv;
  const int e  = t % E_;
  const int b  = t / E_;
  const int row = ei7[(size_t)b * 2 * E_ + e];
  const int col = ei7[(size_t)b * 2 * E_ + E_ + e];
  const float nm = norm[b * E_ + e];
  const float4 s = *(const float4*)(sup + (size_t)(b * N_ + col) * F + f4 * 4);
  float* op = outp + (size_t)(b * N_ + row) * F + f4 * 4;
  atomicAdd(op + 0, nm * s.x);
  atomicAdd(op + 1, nm * s.y);
  atomicAdd(op + 2, nm * s.z);
  atomicAdd(op + 3, nm * s.w);
}

// out1b = bf16(relu(out1 + b1[f]))
__global__ void __launch_bounds__(256) relu_bias_kernel(
    const float* __restrict__ out1, const float* __restrict__ b1,
    __bf16* __restrict__ out1b) {
  const int idx = blockIdx.x * 256 + threadIdx.x;
  const float v = out1[idx] + b1[idx % FH_];
  out1b[idx] = (__bf16)fmaxf(v, 0.f);
}

// support2 = out1b @ W2 : [400,64]x[64,32]; grid (25,16), block 64
__global__ void __launch_bounds__(64) support2_kernel(
    const __bf16* __restrict__ out1b, const __bf16* __restrict__ w2t,
    float* __restrict__ sup2) {
  const int lane = threadIdx.x & 31;
  const int nt   = threadIdx.x >> 5;
  const int mt   = blockIdx.x;
  const int b    = blockIdx.y;
  const __bf16* Ab = out1b + (size_t)(b * N_ + mt * 16) * FH_;
  const __bf16* Bt = w2t   + (size_t)b * FOUT_ * FH_ + (size_t)(nt * 16) * FH_;
  v8f acc = {};
#pragma unroll
  for (int kt = 0; kt < FH_; kt += 32) {
    v16bf a  = load_frag_a_bf16(Ab + kt, FH_, lane);
    v16bf bb = load_frag_b_bf16(Bt + kt, FH_, lane);
    acc = WMMA_BF16(a, bb, acc);
  }
  const int n = lane & 15, mb = (lane >> 4) * 8;
  float* Cp = sup2 + (size_t)(b * N_ + mt * 16) * FOUT_ + nt * 16 + n;
#pragma unroll
  for (int i = 0; i < 8; ++i) Cp[(size_t)(mb + i) * FOUT_] = acc[i];
}

// tr = bf16( (out2 + b2) @ W_fc^T + b_fc )  (b_fc enters via C operand)
__global__ void __launch_bounds__(64) transform_kernel(
    const float* __restrict__ out2, const float* __restrict__ b2,
    const float* __restrict__ Wfc, const float* __restrict__ bfc,
    __bf16* __restrict__ tr) {
  const int lane = threadIdx.x & 31;
  const int nt   = threadIdx.x >> 5;
  const int mt   = blockIdx.x;
  const int b    = blockIdx.y;
  const int r  = lane & 15;
  const int kb = (lane >> 4) * 8;
  const float* ap = out2 + (size_t)(b * N_ + mt * 16 + r) * FOUT_ + kb;
  union { v16bf v; __bf16 e[16]; } ua;
#pragma unroll
  for (int i = 0; i < 8; ++i) ua.e[i]     = (__bf16)(ap[i]      + b2[kb + i]);
#pragma unroll
  for (int i = 0; i < 8; ++i) ua.e[8 + i] = (__bf16)(ap[16 + i] + b2[kb + 16 + i]);
  const int c   = lane & 15;
  const int kbB = (lane >> 4) * 8;
  v16bf bb = cvt16_f32(Wfc + (size_t)(nt * 16 + c) * FOUT_ + kbB);
  const float bias = bfc[nt * 16 + c];
  v8f acc;
#pragma unroll
  for (int i = 0; i < 8; ++i) acc[i] = bias;
  acc = WMMA_BF16(ua.v, bb, acc);
  const int mb = (lane >> 4) * 8;
  __bf16* tp = tr + (size_t)(b * N_ + mt * 16) * FOUT_ + nt * 16 + c;
#pragma unroll
  for (int i = 0; i < 8; ++i) tp[(size_t)(mb + i) * FOUT_] = (__bf16)acc[i];
}

// out[b][n][m] = tanh( sum_d tr[b][n][d] * tr[b][m][d] );  one wave per tile
__global__ void __launch_bounds__(256) gram_kernel(
    const __bf16* __restrict__ tr, float* __restrict__ out) {
  const int lane = threadIdx.x & 31;
  const int wv   = threadIdx.x >> 5;
  const int tile = blockIdx.x * 8 + wv;
  const int b  = tile / 625;
  const int rm = tile % 625;
  const int mt = rm / 25;
  const int nt = rm % 25;
  const __bf16* trb = tr + (size_t)b * N_ * FOUT_;
  v16bf a  = load_frag_a_bf16(trb + (size_t)(mt * 16) * FOUT_, FOUT_, lane);
  v16bf bb = load_frag_b_bf16(trb + (size_t)(nt * 16) * FOUT_, FOUT_, lane);
  v8f acc = {};
  acc = WMMA_BF16(a, bb, acc);
  const int n = lane & 15, mb = (lane >> 4) * 8;
  float* op = out + (size_t)b * N_ * N_ + (size_t)(mt * 16) * N_ + nt * 16 + n;
#pragma unroll
  for (int i = 0; i < 8; ++i) op[(size_t)(mb + i) * N_] = tanhf(acc[i]);
}

// ---------------------------------------------------------------------------
// Host: workspace carving + launch sequence (all on `stream`, capture-safe)
// ---------------------------------------------------------------------------
extern "C" void kernel_launch(void* const* d_in, const int* in_sizes, int n_in,
                              void* d_out, int out_size, void* d_ws, size_t ws_size,
                              hipStream_t stream) {
  (void)in_sizes; (void)n_in; (void)out_size;
  const float* x    = (const float*)d_in[0];
  const float* ew   = (const float*)d_in[1];
  const float* Wih1 = (const float*)d_in[2];
  const float* bih1 = (const float*)d_in[4];
  const float* bhh1 = (const float*)d_in[5];
  const float* Wih2 = (const float*)d_in[6];
  const float* bih2 = (const float*)d_in[8];
  const float* bhh2 = (const float*)d_in[9];
  const float* b1   = (const float*)d_in[10];
  const float* b2   = (const float*)d_in[11];
  const float* Wfc  = (const float*)d_in[12];
  const float* bfc  = (const float*)d_in[13];
  const int*   ei   = (const int*)d_in[14];
  float* out = (float*)d_out;

  char* ws = (char*)d_ws;
  size_t off = 0;
  auto carve = [&](size_t bytes) {
    void* p = ws + off;
    off += (bytes + 255) & ~(size_t)255;
    return p;
  };
  __bf16* h1a  = (__bf16*)carve((size_t)B_ * H1_ * 2);
  __bf16* h1b  = (__bf16*)carve((size_t)B_ * H1_ * 2);
  __bf16* h2a  = (__bf16*)carve((size_t)B_ * H2_ * 2);
  __bf16* h2b  = (__bf16*)carve((size_t)B_ * H2_ * 2);
  __bf16* w1t  = (__bf16*)carve((size_t)B_ * FIN_ * FH_ * 2);
  __bf16* w2t  = (__bf16*)carve((size_t)B_ * FOUT_ * FH_ * 2);
  float*  deg  = (float*)carve((size_t)B_ * N_ * 4);
  float*  norm = (float*)carve((size_t)B_ * E_ * 4);
  float*  sup1 = (float*)carve((size_t)B_ * N_ * FH_ * 4);
  float*  out1 = (float*)carve((size_t)B_ * N_ * FH_ * 4);
  __bf16* out1b= (__bf16*)carve((size_t)B_ * N_ * FH_ * 2);
  float*  sup2 = (float*)carve((size_t)B_ * N_ * FOUT_ * 4);
  float*  out2 = (float*)carve((size_t)B_ * N_ * FOUT_ * 4);
  __bf16* tr   = (__bf16*)carve((size_t)B_ * N_ * FOUT_ * 2);

  // Packed bf16 weight images (101 + 25 MB -> resident in 192 MB L2)
  const size_t w1p_elems = (size_t)3 * (H1_ / 16) * (H1_ / 32) * 512;
  const size_t w2p_elems = (size_t)3 * (H2_ / 16) * (H2_ / 32) * 512;
  __bf16* w1p = (__bf16*)carve(w1p_elems * 2);
  __bf16* w2p = (__bf16*)carve(w2p_elems * 2);
  const bool use_pack = (off <= ws_size);

  hipMemsetAsync(deg,  0, (size_t)B_ * N_ * 4, stream);
  hipMemsetAsync(out1, 0, (size_t)B_ * N_ * FH_ * 4, stream);
  hipMemsetAsync(out2, 0, (size_t)B_ * N_ * FOUT_ * 4, stream);

  // --- GRU weight evolution: step 0 is bias-only, then 7 matmul steps ---
  gru_init_kernel<<<H1_ / 256, 256, 0, stream>>>(bih1, bhh1, h1a, H1_);
  gru_init_kernel<<<H2_ / 256, 256, 0, stream>>>(bih2, bhh2, h2a, H2_);

  __bf16 *c1 = h1a, *n1 = h1b, *c2 = h2a, *n2 = h2b;
  if (use_pack) {
    pack_w_kernel<<<(int)(w1p_elems / 256), 256, 0, stream>>>(Wih1, w1p, H1_);
    pack_w_kernel<<<(int)(w2p_elems / 256), 256, 0, stream>>>(Wih2, w2p, H2_);
    for (int t = 1; t < T_; ++t) {
      gru_step_packed_kernel<<<H1_ / 16, 512, 0, stream>>>(c1, w1p, bih1, bhh1, n1, H1_);
      gru_step_packed_kernel<<<H2_ / 16, 512, 0, stream>>>(c2, w2p, bih2, bhh2, n2, H2_);
      __bf16* tmp;
      tmp = c1; c1 = n1; n1 = tmp;
      tmp = c2; c2 = n2; n2 = tmp;
    }
  } else {
    for (int t = 1; t < T_; ++t) {
      gru_step_kernel<<<H1_ / 16, 256, 0, stream>>>(c1, Wih1, bih1, bhh1, n1, H1_);
      gru_step_kernel<<<H2_ / 16, 256, 0, stream>>>(c2, Wih2, bih2, bhh2, n2, H2_);
      __bf16* tmp;
      tmp = c1; c1 = n1; n1 = tmp;
      tmp = c2; c2 = n2; n2 = tmp;
    }
  }
  // c1 = final h1 [16][4096], c2 = final h2 [16][2048]

  transposeW_kernel<<<(B_ * H1_ + B_ * H2_) / 256, 256, 0, stream>>>(c1, c2, w1t, w2t);

  // --- only t = T-1 contributes to the output ---
  const float* x7  = x  + (size_t)(T_ - 1) * B_ * N_ * FIN_;
  const int*   ei7 = ei + (size_t)(T_ - 1) * B_ * 2 * E_;
  const float* ew7 = ew + (size_t)(T_ - 1) * B_ * E_;

  deg_kernel <<<(B_ * E_) / 256, 256, 0, stream>>>(ei7, deg);
  norm_kernel<<<(B_ * E_) / 256, 256, 0, stream>>>(ei7, ew7, deg, norm);

  support1_kernel<<<dim3(N_ / 16, B_), 128, 0, stream>>>(x7, w1t, sup1);
  scatter_kernel<<<(B_ * E_ * (FH_ / 4)) / 256, 256, 0, stream>>>(sup1, norm, ei7, out1, FH_);
  relu_bias_kernel<<<(B_ * N_ * FH_) / 256, 256, 0, stream>>>(out1, b1, out1b);

  support2_kernel<<<dim3(N_ / 16, B_), 64, 0, stream>>>(out1b, w2t, sup2);
  scatter_kernel<<<(B_ * E_ * (FOUT_ / 4)) / 256, 256, 0, stream>>>(sup2, norm, ei7, out2, FOUT_);

  transform_kernel<<<dim3(N_ / 16, B_), 64, 0, stream>>>(out2, b2, Wfc, bfc, tr);

  gram_kernel<<<(B_ * (N_ / 16) * (N_ / 16)) / 8, 256, 0, stream>>>(tr, out);
}
